// VQVAE2_38147899523234
// MI455X (gfx1250) — matrix operand order repaired and checked by reading, not compiled, not run on hardware
//
#include <hip/hip_runtime.h>

typedef _Float16 half_t;
typedef __attribute__((ext_vector_type(16))) _Float16 v16h;
typedef __attribute__((ext_vector_type(8)))  _Float16 v8h;
typedef __attribute__((ext_vector_type(8)))  float    v8f;

#define TM 128
#define TN 128
#define TK 32
#define ASTR 40   // f16 elements per LDS row (padded, 80B, 16B-aligned rows)
#define BSTR 40

// ---------------------------------------------------------------------------
// Generic implicit-GEMM convolution (mode 0) / ConvTranspose2d k4s2p1 (mode 1)
// D = A(weights, 128xKp f16 row-major, rows >= M zero-padded) x B(im2col of X)
//     [+ bias][+ residual][ReLU]
// A tile is staged with CDNA5 async global->LDS DMA (ASYNCcnt).
// ---------------------------------------------------------------------------
__global__ __launch_bounds__(256) void conv_gemm_wmma(
    const half_t* __restrict__ A, const float* __restrict__ bias,
    const half_t* __restrict__ X, const half_t* __restrict__ R,
    half_t* __restrict__ Y, float* __restrict__ Yf,
    int M, int Kp, int Ktot,
    int Nb, int Cin, int Hin, int Win, int Hout, int Wout,
    int KH, int KW, int stride, int pad, int mode, int relu)
{
    __shared__ __align__(16) half_t As[TM * ASTR];
    __shared__ __align__(16) half_t Bs[TN * BSTR];

    const int tid  = threadIdx.x;
    const int lane = tid & 31;
    const int wave = tid >> 5;
    const int l15  = lane & 15;
    const int hsel = lane >> 4;           // 0: lanes 0-15, 1: lanes 16-31
    const int wm   = wave & 1;            // 64-row half of the 128-row tile
    const int wn   = wave >> 1;           // 32-col quarter of the 128-col tile

    const long long Ncols = (long long)Nb * Hout * Wout;
    const int  m0   = blockIdx.y * TM;
    const long long col0 = (long long)blockIdx.x * TN;

    v8f acc[4][2];
    for (int i = 0; i < 4; ++i)
        for (int j = 0; j < 2; ++j)
            acc[i][j] = {};

    const int arow  = tid >> 1;          // 0..127: tile row (A) / tile col (B)
    const int ahalf = (tid & 1) * 16;    // 0 or 16 within the 32-deep K step

    // Async A staging addresses (A is padded to 128 rows -> always in range)
    const half_t* asrc0 = A + (long long)(m0 + arow) * Kp + ahalf;
    const unsigned lds_a = (unsigned)(uintptr_t)&As[arow * ASTR + ahalf];

    // Pre-decompose this thread's B column
    const long long ng = col0 + arow;
    const bool ncol_ok = (ng < Ncols);
    int bN = 0, ho = 0, wo = 0;
    if (ncol_ok) {
        wo = (int)(ng % Wout);
        long long t = ng / Wout;
        ho = (int)(t % Hout);
        bN = (int)(t / Hout);
    }
    const int khw = KH * KW;
    const int HW  = Hin * Win;
    const long long xbase = (long long)bN * Cin * HW;

    for (int kc = 0; kc < Kp; kc += TK) {
        __syncthreads();
        // ---- stage A tile: async global -> LDS DMA (32B per lane) ----
        {
            const unsigned long long ga = (unsigned long long)(uintptr_t)(asrc0 + kc);
            asm volatile(
                "global_load_async_to_lds_b128 %0, %1, off\n\t"
                "global_load_async_to_lds_b128 %0, %1, off offset:16"
                :: "v"(lds_a), "v"(ga) : "memory");
        }
        // ---- stage B tile transposed Bt[n][k]: im2col gather ----
        {
            const int k0 = kc + ahalf;
            int ci, kh, kw;
            {
                int t = k0;
                ci = t / khw; t -= ci * khw;
                kh = t / KW;  kw = t - kh * KW;
            }
            half_t vals[16];
            #pragma unroll
            for (int i = 0; i < 16; ++i) {
                half_t v = (half_t)0.f;
                if (ncol_ok && (k0 + i) < Ktot) {
                    int hi2, wi2; bool ok;
                    if (mode == 0) {
                        hi2 = ho * stride - pad + kh;
                        wi2 = wo * stride - pad + kw;
                        ok = (hi2 >= 0) & (hi2 < Hin) & (wi2 >= 0) & (wi2 < Win);
                    } else { // ConvTranspose2d
                        const int th = ho + pad - kh;
                        const int tw = wo + pad - kw;
                        ok = (th >= 0) & (tw >= 0) &&
                             ((th % stride) == 0) && ((tw % stride) == 0);
                        hi2 = th / stride; wi2 = tw / stride;
                        ok = ok && (hi2 < Hin) && (wi2 < Win);
                    }
                    if (ok)
                        v = X[xbase + (long long)ci * HW + hi2 * Win + wi2];
                }
                vals[i] = v;
                // incremental (ci,kh,kw) update — no per-element div/mod
                if (++kw == KW) { kw = 0; if (++kh == KH) { kh = 0; ++ci; } }
            }
            #pragma unroll
            for (int i = 0; i < 16; ++i)
                Bs[arow * BSTR + ahalf + i] = vals[i];
        }
        asm volatile("s_wait_asynccnt 0x0" ::: "memory");
        __syncthreads();

        // ---- fragments + 8 WMMA per K step ----
        const int ka = hsel ? 8 : 0;     // A lane K split per CDNA5 layout
        v16h afr[4];
        #pragma unroll
        for (int i = 0; i < 4; ++i) {
            const int r = wm * 64 + i * 16 + l15;
            v8h lo = *(v8h*)&As[r * ASTR + ka];
            v8h hi = *(v8h*)&As[r * ASTR + ka + 16];
            #pragma unroll
            for (int e = 0; e < 8; ++e) { afr[i][e] = lo[e]; afr[i][e + 8] = hi[e]; }
        }
        v16h bfr[2];
        #pragma unroll
        for (int j = 0; j < 2; ++j) {
            const int c = wn * 32 + j * 16 + l15;
            v8h lo = *(v8h*)&Bs[c * BSTR + hsel * 8];
            v8h hi = *(v8h*)&Bs[c * BSTR + hsel * 8 + 16];
            #pragma unroll
            for (int e = 0; e < 8; ++e) { bfr[j][e] = lo[e]; bfr[j][e + 8] = hi[e]; }
        }
        #pragma unroll
        for (int i = 0; i < 4; ++i)
            #pragma unroll
            for (int j = 0; j < 2; ++j)
                acc[i][j] = __builtin_amdgcn_wmma_f32_16x16x32_f16(
                    false, afr[i], false, bfr[j], (short)0, acc[i][j], false, false);
    }

    // ---- epilogue: bias + residual + ReLU, f16 and/or f32 store ----
    #pragma unroll
    for (int i = 0; i < 4; ++i) {
        const int mb = m0 + wm * 64 + i * 16 + hsel * 8;
        #pragma unroll
        for (int j = 0; j < 2; ++j) {
            const long long nb_ = col0 + wn * 32 + j * 16 + l15;
            if (nb_ >= Ncols) continue;
            const int owo = (int)(nb_ % Wout);
            long long t = nb_ / Wout;
            const int oho = (int)(t % Hout);
            const int ob  = (int)(t / Hout);
            #pragma unroll
            for (int r = 0; r < 8; ++r) {
                const int mg = mb + r;
                if (mg >= M) break;
                float v = acc[i][j][r];
                if (bias) v += bias[mg];
                const long long oidx =
                    (((long long)ob * M + mg) * Hout + oho) * Wout + owo;
                if (R) v += (float)R[oidx];
                if (relu) v = v > 0.f ? v : 0.f;
                if (Y)  Y[oidx]  = (half_t)v;
                if (Yf) Yf[oidx] = v;
            }
        }
    }
}

// ---------------------------------------------------------------------------
// Vector quantization: in-place nearest-code replace + loss accumulation.
// Rows are 64 contiguous f16 (faithful to reference's raw NCHW reshape).
// ---------------------------------------------------------------------------
__global__ __launch_bounds__(256) void vq_kernel(
    half_t* __restrict__ z, const float* __restrict__ emb,
    float* __restrict__ acc, int rows)
{
    __shared__ half_t cb[512 * 64];                 // 64KB codebook in LDS
    for (int i = threadIdx.x; i < 512 * 64; i += 256)
        cb[i] = (half_t)emb[i];
    __syncthreads();

    const int row = blockIdx.x * blockDim.x + threadIdx.x;
    float lsum = 0.f;
    if (row < rows) {
        half_t* zp = z + (long long)row * 64;
        float zr[64];
        #pragma unroll
        for (int i = 0; i < 64; ++i) zr[i] = (float)zp[i];

        float best = 3.4e38f;
        int   bidx = 0;
        #pragma unroll 1
        for (int c = 0; c < 512; ++c) {
            const half_t* e = &cb[c * 64];
            float d = 0.f;
            #pragma unroll 8
            for (int i = 0; i < 64; ++i) {
                const float t = zr[i] - (float)e[i];
                d = fmaf(t, t, d);
            }
            if (d < best) { best = d; bidx = c; }
        }
        lsum = best;                                 // ||z - e_best||^2
        const half_t* e = &cb[bidx * 64];
        #pragma unroll
        for (int i = 0; i < 64; ++i) zp[i] = e[i];   // straight-through value
    }
    for (int off = 16; off > 0; off >>= 1)
        lsum += __shfl_down(lsum, off, 32);
    if ((threadIdx.x & 31) == 0) atomicAdd(acc, lsum);
}

// ---------------------------------------------------------------------------
// Elementwise / conversion helpers
// ---------------------------------------------------------------------------
__global__ void cvt_f32_f16(const float* __restrict__ src,
                            half_t* __restrict__ dst, int n)
{
    int i = blockIdx.x * blockDim.x + threadIdx.x;
    if (i < n) dst[i] = (half_t)src[i];
}

// Conv weight OIHW -> A[Mp][Kp] (natural flatten, zero pad rows/cols)
__global__ void cvt_w_conv(const float* __restrict__ w, half_t* __restrict__ A,
                           int M, int Mp, int K, int Kp)
{
    int i = blockIdx.x * blockDim.x + threadIdx.x;
    if (i >= Mp * Kp) return;
    int m = i / Kp, k = i - m * Kp;
    A[i] = (m < M && k < K) ? (half_t)w[(long long)m * K + k] : (half_t)0.f;
}

// ConvT weight (Cin,Cout,4,4) -> A[co][ (ci,kh,kw) ], zero pad
__global__ void cvt_w_convT(const float* __restrict__ w, half_t* __restrict__ A,
                            int Cin, int Cout, int Mp, int Kp)
{
    int i = blockIdx.x * blockDim.x + threadIdx.x;
    if (i >= Mp * Kp) return;
    int co = i / Kp, k = i - co * Kp;
    half_t v = (half_t)0.f;
    if (co < Cout && k < Cin * 16) {
        int ci = k >> 4, khw = k & 15;
        v = (half_t)w[((long long)ci * Cout + co) * 16 + khw];
    }
    A[i] = v;
}

// nearest-downsample(256->128) of d into first 128 channels + concat q_b
__global__ void ds_concat(const half_t* __restrict__ d,
                          const half_t* __restrict__ qb,
                          half_t* __restrict__ comb, int total)
{
    int i = blockIdx.x * blockDim.x + threadIdx.x;
    if (i >= total) return;
    int w = i & 127; int t = i >> 7;
    int h = t & 127; t >>= 7;
    int c = t % 192; int n = t / 192;
    half_t v;
    if (c < 128)
        v = d[(((long long)n * 128 + c) * 256 + 2 * h) * 256 + 2 * w];
    else
        v = qb[(((long long)n * 64 + (c - 128)) * 128 + h) * 128 + w];
    comb[i] = v;
}

__global__ void recon_loss(const float* __restrict__ xr,
                           const float* __restrict__ x,
                           float* __restrict__ acc, int n)
{
    int i = blockIdx.x * blockDim.x + threadIdx.x;
    float d = 0.f;
    if (i < n) { float t = xr[i] - x[i]; d = t * t; }
    for (int off = 16; off > 0; off >>= 1)
        d += __shfl_down(d, off, 32);
    if ((threadIdx.x & 31) == 0) atomicAdd(acc, d);
}

__global__ void zero_accs(float* a) { if (threadIdx.x < 3) a[threadIdx.x] = 0.f; }

__global__ void finalize(const float* __restrict__ a, float* __restrict__ out0,
                         float rn, float vn)
{
    // total = recon_mse + 1.25*mean||q_t-z_t||^2 + 1.25*mean||q_b-z_b||^2
    out0[0] = a[2] / rn + 1.25f * (a[0] / vn) + 1.25f * (a[1] / vn);
}

// ---------------------------------------------------------------------------
// Host side
// ---------------------------------------------------------------------------
extern "C" void kernel_launch(void* const* d_in, const int* in_sizes, int n_in,
                              void* d_out, int out_size, void* d_ws, size_t ws_size,
                              hipStream_t stream) {
    (void)in_sizes; (void)n_in; (void)out_size; (void)ws_size;

    // Input order (flattened setup_inputs dict):
    const float* x       = (const float*)d_in[0];
    const float* et_c1w  = (const float*)d_in[1];
    const float* et_c1b  = (const float*)d_in[2];
    const float* et_rw1  = (const float*)d_in[3];
    const float* et_rb1  = (const float*)d_in[4];
    const float* et_rw2  = (const float*)d_in[5];
    const float* et_rb2  = (const float*)d_in[6];
    const float* et_c2w  = (const float*)d_in[7];
    const float* et_c2b  = (const float*)d_in[8];
    const float* eb_c1w  = (const float*)d_in[9];
    const float* eb_c1b  = (const float*)d_in[10];
    const float* eb_rw1  = (const float*)d_in[11];
    const float* eb_rb1  = (const float*)d_in[12];
    const float* eb_rw2  = (const float*)d_in[13];
    const float* eb_rb2  = (const float*)d_in[14];
    const float* eb_c2w  = (const float*)d_in[15];
    const float* eb_c2b  = (const float*)d_in[16];
    const float* emb_top = (const float*)d_in[17];
    const float* emb_bot = (const float*)d_in[18];
    const float* dt_tw   = (const float*)d_in[19];
    const float* dt_tb   = (const float*)d_in[20];
    const float* dt_rw1  = (const float*)d_in[21];
    const float* dt_rb1  = (const float*)d_in[22];
    const float* dt_rw2  = (const float*)d_in[23];
    const float* dt_rb2  = (const float*)d_in[24];
    const float* db_tw   = (const float*)d_in[25];
    const float* db_tb   = (const float*)d_in[26];
    const float* db_rw1  = (const float*)d_in[27];
    const float* db_rb1  = (const float*)d_in[28];
    const float* db_rw2  = (const float*)d_in[29];
    const float* db_rb2  = (const float*)d_in[30];
    const float* db_ow   = (const float*)d_in[31];
    const float* db_ob   = (const float*)d_in[32];

    float* out = (float*)d_out;          // out[0]=loss, out[1..]=x_recon (f32)

    // ---- workspace arena ----
    char* ws = (char*)d_ws;
    size_t off = 0;
    auto alloc = [&](size_t bytes) -> char* {
        char* p = ws + off;
        off = (off + bytes + 255) & ~(size_t)255;
        return p;
    };
    const long long S_X = 8LL * 3 * 256 * 256;          // 1572864
    const long long S_T = 8LL * 128 * 128 * 128;        // 16777216
    const long long S_Z = 8LL * 64 * 128 * 128;         // 8388608
    const long long S_D = 8LL * 128 * 256 * 256;        // 67108864
    const long long S_C = 8LL * 192 * 128 * 128;        // 25165824

    half_t* XF   = (half_t*)alloc(S_X * 2);
    half_t* T0   = (half_t*)alloc(S_T * 2);
    half_t* T1   = (half_t*)alloc(S_T * 2);
    half_t* ZT   = (half_t*)alloc(S_Z * 2);
    half_t* ZB   = (half_t*)alloc(S_Z * 2);
    half_t* D0   = (half_t*)alloc(S_D * 2);
    half_t* D1   = (half_t*)alloc(S_D * 2);
    half_t* COMB = (half_t*)alloc(S_C * 2);

    // All A buffers padded to 128 rows so async staging never goes OOB.
    half_t* Aet1 = (half_t*)alloc(128 * 64   * 2);
    half_t* Aetr1= (half_t*)alloc(128 * 1152 * 2);
    half_t* Aetr2= (half_t*)alloc(128 * 1152 * 2);
    half_t* Aet2 = (half_t*)alloc(128 * 128  * 2);
    half_t* Aeb1 = (half_t*)alloc(128 * 64   * 2);
    half_t* Aebr1= (half_t*)alloc(128 * 1152 * 2);
    half_t* Aebr2= (half_t*)alloc(128 * 1152 * 2);
    half_t* Aeb2 = (half_t*)alloc(128 * 128  * 2);
    half_t* Adt  = (half_t*)alloc(128 * 1024 * 2);
    half_t* Adtr1= (half_t*)alloc(128 * 1152 * 2);
    half_t* Adtr2= (half_t*)alloc(128 * 1152 * 2);
    half_t* Adb  = (half_t*)alloc(128 * 3072 * 2);
    half_t* Adbr1= (half_t*)alloc(128 * 1152 * 2);
    half_t* Adbr2= (half_t*)alloc(128 * 1152 * 2);
    half_t* Aow  = (half_t*)alloc(128 * 128  * 2);
    float*  ACC  = (float*)alloc(3 * 4);            // [vq_top, vq_bot, recon]

    auto cdiv = [](long long n, int b) { return (unsigned)((n + b - 1) / b); };

    zero_accs<<<1, 32, 0, stream>>>(ACC);
    cvt_f32_f16<<<cdiv(S_X, 256), 256, 0, stream>>>(x, XF, (int)S_X);

    // weight conversions (row-padded to 128)
    cvt_w_conv <<<cdiv(128 * 64,   256), 256, 0, stream>>>(et_c1w, Aet1,  128, 128, 48,   64);
    cvt_w_conv <<<cdiv(128 * 1152, 256), 256, 0, stream>>>(et_rw1, Aetr1, 128, 128, 1152, 1152);
    cvt_w_conv <<<cdiv(128 * 1152, 256), 256, 0, stream>>>(et_rw2, Aetr2, 128, 128, 1152, 1152);
    cvt_w_conv <<<cdiv(128 * 128,  256), 256, 0, stream>>>(et_c2w, Aet2,  64,  128, 128,  128);
    cvt_w_conv <<<cdiv(128 * 64,   256), 256, 0, stream>>>(eb_c1w, Aeb1,  128, 128, 48,   64);
    cvt_w_conv <<<cdiv(128 * 1152, 256), 256, 0, stream>>>(eb_rw1, Aebr1, 128, 128, 1152, 1152);
    cvt_w_conv <<<cdiv(128 * 1152, 256), 256, 0, stream>>>(eb_rw2, Aebr2, 128, 128, 1152, 1152);
    cvt_w_conv <<<cdiv(128 * 128,  256), 256, 0, stream>>>(eb_c2w, Aeb2,  64,  128, 128,  128);
    cvt_w_convT<<<cdiv(128 * 1024, 256), 256, 0, stream>>>(dt_tw,  Adt,   64,  128, 128, 1024);
    cvt_w_conv <<<cdiv(128 * 1152, 256), 256, 0, stream>>>(dt_rw1, Adtr1, 128, 128, 1152, 1152);
    cvt_w_conv <<<cdiv(128 * 1152, 256), 256, 0, stream>>>(dt_rw2, Adtr2, 128, 128, 1152, 1152);
    cvt_w_convT<<<cdiv(128 * 3072, 256), 256, 0, stream>>>(db_tw,  Adb,   192, 128, 128, 3072);
    cvt_w_conv <<<cdiv(128 * 1152, 256), 256, 0, stream>>>(db_rw1, Adbr1, 128, 128, 1152, 1152);
    cvt_w_conv <<<cdiv(128 * 1152, 256), 256, 0, stream>>>(db_rw2, Adbr2, 128, 128, 1152, 1152);
    cvt_w_conv <<<cdiv(128 * 128,  256), 256, 0, stream>>>(db_ow,  Aow,   3,   128, 128,  128);

    auto conv = [&](const half_t* A, const float* bias, const half_t* Xin,
                    const half_t* Rres, half_t* Y, float* Yf,
                    int M, int Ktot, int Kp, int Cin, int Hin, int Win,
                    int Hout, int Wout, int KH, int KW, int s, int p,
                    int mode, int relu) {
        long long Ncols = 8LL * Hout * Wout;
        dim3 g(cdiv(Ncols, TN), cdiv(M, TM));
        conv_gemm_wmma<<<g, 256, 0, stream>>>(A, bias, Xin, Rres, Y, Yf,
            M, Kp, Ktot, 8, Cin, Hin, Win, Hout, Wout, KH, KW, s, p, mode, relu);
    };

    // ===== encoder top =====
    conv(Aet1,  et_c1b, XF, nullptr, T0, nullptr, 128, 48,   64,   3,  256,256, 128,128, 4,4, 2,1, 0, 0);
    conv(Aetr1, et_rb1, T0, nullptr, T1, nullptr, 128, 1152, 1152, 128,128,128, 128,128, 3,3, 1,1, 0, 1);
    conv(Aetr2, et_rb2, T1, T0,      T0, nullptr, 128, 1152, 1152, 128,128,128, 128,128, 3,3, 1,1, 0, 1);
    conv(Aet2,  et_c2b, T0, nullptr, ZT, nullptr, 64,  128,  128,  128,128,128, 128,128, 1,1, 1,0, 0, 0);

    // ===== encoder bottom =====
    conv(Aeb1,  eb_c1b, XF, nullptr, T0, nullptr, 128, 48,   64,   3,  256,256, 128,128, 4,4, 2,1, 0, 0);
    conv(Aebr1, eb_rb1, T0, nullptr, T1, nullptr, 128, 1152, 1152, 128,128,128, 128,128, 3,3, 1,1, 0, 1);
    conv(Aebr2, eb_rb2, T1, T0,      T0, nullptr, 128, 1152, 1152, 128,128,128, 128,128, 3,3, 1,1, 0, 1);
    conv(Aeb2,  eb_c2b, T0, nullptr, ZB, nullptr, 64,  128,  128,  128,128,128, 128,128, 1,1, 1,0, 0, 0);

    // ===== vector quantization (in place: ZT/ZB become q_t/q_b) =====
    vq_kernel<<<cdiv(S_Z / 64, 256), 256, 0, stream>>>(ZT, emb_top, ACC + 0, (int)(S_Z / 64));
    vq_kernel<<<cdiv(S_Z / 64, 256), 256, 0, stream>>>(ZB, emb_bot, ACC + 1, (int)(S_Z / 64));

    // ===== decoder top: convT -> resblock -> (relu) =====
    conv(Adt,   dt_tb,  ZT, nullptr, D0, nullptr, 128, 1024, 1024, 64, 128,128, 256,256, 4,4, 2,1, 1, 0);
    conv(Adtr1, dt_rb1, D0, nullptr, D1, nullptr, 128, 1152, 1152, 128,256,256, 256,256, 3,3, 1,1, 0, 1);
    conv(Adtr2, dt_rb2, D1, D0,      D0, nullptr, 128, 1152, 1152, 128,256,256, 256,256, 3,3, 1,1, 0, 1);

    // nearest downsample 256->128 + concat with q_b -> COMB (192ch,128x128)
    ds_concat<<<cdiv(S_C, 256), 256, 0, stream>>>(D0, ZB, COMB, (int)S_C);

    // ===== decoder bottom =====
    conv(Adb,   db_tb,  COMB, nullptr, D0, nullptr, 128, 3072, 3072, 192,128,128, 256,256, 4,4, 2,1, 1, 0);
    conv(Adbr1, db_rb1, D0,  nullptr, D1, nullptr, 128, 1152, 1152, 128,256,256, 256,256, 3,3, 1,1, 0, 1);
    conv(Adbr2, db_rb2, D1,  D0,      D0, nullptr, 128, 1152, 1152, 128,256,256, 256,256, 3,3, 1,1, 0, 1);
    // final 1x1 projection, f32 straight into d_out[1..]
    conv(Aow,   db_ob,  D0,  nullptr, nullptr, out + 1, 3, 128, 128, 128,256,256, 256,256, 1,1, 1,0, 0, 0);

    // ===== losses =====
    recon_loss<<<cdiv(S_X, 256), 256, 0, stream>>>(out + 1, x, ACC + 2, (int)S_X);
    finalize<<<1, 1, 0, stream>>>(ACC, out, (float)S_X, (float)S_Z);
}
